// AtomicPotentialModel_72335839200044
// MI455X (gfx1250) — matrix operand order repaired and checked by reading, not compiled, not run on hardware
//
#include <hip/hip_runtime.h>

#define NB     4
#define NATOMS 1250
#define NPROBES 50000
#define NEDGES 125000
#define TT     3
#define DD     128
#define EDGETOT (NB*NEDGES)   /* 500000 */
#define PTOT    (NB*NPROBES)  /* 200000 */

typedef _Float16 h16;
typedef _Float16 v16h __attribute__((ext_vector_type(16)));
typedef float    v8f  __attribute__((ext_vector_type(8)));

union V16H {
    v16h   v;
    h16    h[16];
    float4 f4[2];
};

__global__ void __launch_bounds__(256) zero_kernel(float* p, long n) {
    long i = (long)blockIdx.x * blockDim.x + threadIdx.x;
    long stride = (long)gridDim.x * blockDim.x;
    for (; i < n; i += stride) p[i] = 0.0f;
}

__global__ void __launch_bounds__(128) edge_kernel(
    const float* __restrict__ atom_xyz,   // [B*NATOMS*3]
    const float* __restrict__ probe_xyz,  // [B*NPROBES*3]
    const int*   __restrict__ probe_edges,// [B*NEDGES*2]
    const float* __restrict__ ped,        // [B*NEDGES*3]
    const float* __restrict__ cell,       // [B*9]
    const float* __restrict__ atom_rep,   // [T*B*NATOMS*DD]
    const float* __restrict__ w1,         // [T*10*100]
    const float* __restrict__ w2,         // [T*100*16]
    float*       __restrict__ probes)     // [B*NPROBES*DD] accumulated
{
    // B-fragment-ready f16 weights, fully zero-padded so ALL lanes load
    // unconditionally (no EXEC divergence around WMMA fragment loads):
    //   w1T[t][col(112)][k(32)]  (k>=10 or col>=100 -> 0)
    //   w2T[t][col(16)][k(136)]  (k>=100 -> 0)
    __shared__ __align__(16) h16  w1T[3 * 112 * 32];
    __shared__ __align__(16) h16  w2T[3 * 16 * 136];
    __shared__ __align__(16) h16  hbuf[4][16 * 128];   // per-wave h (A2 staging)
    __shared__ float shbuf[4][16 * 17];                 // per-wave sh
    __shared__ float d2buf[4][16 * 17];                 // per-wave D2 tile
    __shared__ float scalbuf[4][16 * 3];                // per-wave scal[edge][t]

    const int tid = threadIdx.x;

    // ---- preload weights (f16, transposed, zero-padded) ----
    for (int i = tid; i < 3 * 112 * 32; i += 128) {
        int t = i / (112 * 32); int r = i % (112 * 32);
        int col = r / 32, k = r % 32;
        float v = (col < 100 && k < 10) ? w1[(t * 10 + k) * 100 + col] : 0.0f;
        w1T[i] = (h16)v;
    }
    for (int i = tid; i < 3 * 16 * 136; i += 128) {
        int t = i / (16 * 136); int r = i % (16 * 136);
        int col = r / 136, k = r % 136;
        float v = (k < 100) ? w2[(t * 100 + k) * 16 + col] : 0.0f;
        w2T[i] = (h16)v;
    }
    // zero h staging once (cols 112..127 are never overwritten; avoids NaN garbage)
    for (int i = tid; i < 4 * 16 * 128; i += 128) ((h16*)hbuf)[i] = (h16)0.0f;
    __syncthreads();

    const int wave = tid >> 5;
    const int lane = tid & 31;
    const int sub  = lane & 15;
    const int hi   = lane >> 4;

    const int tile  = blockIdx.x * 4 + wave;
    const int e     = tile * 16 + sub;      // lane pair (sub, sub+16) share an edge
    const int valid = (e < EDGETOT) ? 1 : 0;
    const int ec    = valid ? e : 0;

    const int b  = ec / NEDGES;
    const int le = ec % NEDGES;
    const int srcl = probe_edges[(b * NEDGES + le) * 2 + 0];
    const int dstl = probe_edges[(b * NEDGES + le) * 2 + 1];
    const int srcg = b * NATOMS + srcl;
    const int dstg = b * NPROBES + dstl;

    // displacement = ped @ cell[b]
    const float p0 = ped[(b * NEDGES + le) * 3 + 0];
    const float p1 = ped[(b * NEDGES + le) * 3 + 1];
    const float p2 = ped[(b * NEDGES + le) * 3 + 2];
    const float* cb = cell + b * 9;
    float dx = p0 * cb[0] + p1 * cb[3] + p2 * cb[6];
    float dy = p0 * cb[1] + p1 * cb[4] + p2 * cb[7];
    float dz = p0 * cb[2] + p1 * cb[5] + p2 * cb[8];

    float vx = probe_xyz[dstg * 3 + 0] - (atom_xyz[srcg * 3 + 0] + dx);
    float vy = probe_xyz[dstg * 3 + 1] - (atom_xyz[srcg * 3 + 1] + dy);
    float vz = probe_xyz[dstg * 3 + 2] - (atom_xyz[srcg * 3 + 2] + dz);

    float r = sqrtf(vx * vx + vy * vy + vz * vz);
    float inv = __builtin_amdgcn_rcpf(fmaxf(r, 1e-9f));   // v_rcp_f32
    float x = vx * inv, y = vy * inv, z = vz * inv;
    float x2 = x * x, y2 = y * y, z2 = z * z;

    // spherical harmonics l=0..3 (reference constants)
    float sh[16];
    sh[0]  = 1.0f;
    sh[1]  = 1.7320508075688772f * x;
    sh[2]  = 1.7320508075688772f * y;
    sh[3]  = 1.7320508075688772f * z;
    sh[4]  = 3.872983346207417f  * x * y;
    sh[5]  = 3.872983346207417f  * y * z;
    sh[6]  = 1.118033988749895f  * (3.0f * z2 - 1.0f);
    sh[7]  = 3.872983346207417f  * x * z;
    sh[8]  = 1.9364916731037085f * (x2 - y2);
    sh[9]  = 2.091650066335189f  * y * (3.0f * x2 - y2);
    sh[10] = 10.246950765959598f * x * y * z;
    sh[11] = 1.6201851746019651f * y * (5.0f * z2 - 1.0f);
    sh[12] = 1.3228756555322954f * (5.0f * z2 * z - 3.0f * z);
    sh[13] = 1.6201851746019651f * x * (5.0f * z2 - 1.0f);
    sh[14] = 5.123475382979799f  * z * (x2 - y2);
    sh[15] = 2.091650066335189f  * x * (x2 - 3.0f * y2);

    // radial basis (10 gaussians)
    float rb[10];
    const float step = 4.0f / 9.0f;
    #pragma unroll
    for (int k = 0; k < 10; k++) {
        float d = (r - step * (float)k) / step;
        rb[k] = __expf(-d * d) * 1.12f;
    }

    if (!hi) {
        #pragma unroll
        for (int n = 0; n < 16; n++) shbuf[wave][sub * 17 + n] = sh[n];
    }
    __syncthreads();

    // ---- A1 fragment built in registers (16-bit A 16x32 layout) ----
    // low lanes: elems 0..7 = K0..7 (rb0..7), elems 8..15 = K16..23 (zero)
    // high lanes: elems 0..7 = K8..15 (rb8,rb9,0..), elems 8..15 = K24..31 (zero)
    V16H a1;
    #pragma unroll
    for (int i = 0; i < 16; i++) a1.h[i] = (h16)0.0f;
    if (!hi) {
        #pragma unroll
        for (int i = 0; i < 8; i++) a1.h[i] = (h16)rb[i];
    } else {
        a1.h[0] = (h16)rb[8];
        a1.h[1] = (h16)rb[9];
    }

    for (int t = 0; t < 3; t++) {
        // ---- GEMM1: [16 edges x 32K] x [32K x 112] ----
        // load all 7 B fragments first (unconditional), then chain WMMAs
        V16H b1f[7];
        #pragma unroll
        for (int j = 0; j < 7; j++) {
            const h16* p = &w1T[(t * 112 + (16 * j + sub)) * 32 + 8 * hi];
            b1f[j].f4[0] = *(const float4*)p;
            b1f[j].f4[1] = *(const float4*)(p + 16);
        }
        v8f acc1[7];
        #pragma unroll
        for (int j = 0; j < 7; j++) {
            v8f zc = {0.f, 0.f, 0.f, 0.f, 0.f, 0.f, 0.f, 0.f};
            acc1[j] = __builtin_amdgcn_wmma_f32_16x16x32_f16(
                false, a1.v, false, b1f[j].v, (short)0, zc, false, false);
        }
        // silu (x * rcp(1+exp(-x)), single v_rcp_f32) + stage h (f16) into LDS
        #pragma unroll
        for (int j = 0; j < 7; j++) {
            #pragma unroll
            for (int i = 0; i < 8; i++) {
                float xx = acc1[j][i];
                float s  = xx * __builtin_amdgcn_rcpf(1.0f + __expf(-xx));
                hbuf[wave][(8 * hi + i) * 128 + 16 * j + sub] = (h16)s;
            }
        }
        __syncthreads();

        // ---- GEMM2: [16 edges x 128K] x [128K x 16] ----
        V16H a2f[4], b2f[4];
        #pragma unroll
        for (int c = 0; c < 4; c++) {
            const h16* pa = &hbuf[wave][sub * 128 + 32 * c + 8 * hi];
            a2f[c].f4[0] = *(const float4*)pa;          // K = 32c + 8hi .. +7
            a2f[c].f4[1] = *(const float4*)(pa + 16);   // K = 32c + 16 + 8hi .. +7
            const h16* pb = &w2T[(t * 16 + sub) * 136 + 32 * c + 8 * hi];
            b2f[c].f4[0] = *(const float4*)pb;
            b2f[c].f4[1] = *(const float4*)(pb + 16);
        }
        v8f acc2 = {0.f, 0.f, 0.f, 0.f, 0.f, 0.f, 0.f, 0.f};
        #pragma unroll
        for (int c = 0; c < 4; c++) {
            acc2 = __builtin_amdgcn_wmma_f32_16x16x32_f16(
                false, a2f[c].v, false, b2f[c].v, (short)0, acc2, false, false);
        }
        // stage D2 and reduce against sh per edge
        #pragma unroll
        for (int i = 0; i < 8; i++)
            d2buf[wave][(8 * hi + i) * 17 + sub] = acc2[i];
        __syncthreads();

        if (!hi) {
            float s = 0.0f;
            #pragma unroll
            for (int n = 0; n < 16; n++)
                s += d2buf[wave][sub * 17 + n] * shbuf[wave][sub * 17 + n];
            scalbuf[wave][sub * 3 + t] = s;
        }
        __syncthreads();
    }

    // ---- fused scatter: probes[dst] += (sum_t scal_t * rep_t[src]) * inv_sqrt_nn ----
    const float INV = 0.22360679774997896f;  // 1/sqrt(20)
    const float* rep0 = atom_rep;
    const float* rep1 = atom_rep + (size_t)(NB * NATOMS) * DD;
    const float* rep2 = atom_rep + (size_t)2 * (NB * NATOMS) * DD;

    for (int q = 0; q < 16; q++) {
        int sg = __shfl(srcg, q, 32);
        int dg = __shfl(dstg, q, 32);
        int vv = __shfl(valid, q, 32);
        if (vv) {
            float s0 = scalbuf[wave][q * 3 + 0];
            float s1 = scalbuf[wave][q * 3 + 1];
            float s2 = scalbuf[wave][q * 3 + 2];
            const float4 a0  = *(const float4*)&rep0[(size_t)sg * DD + lane * 4];
            const float4 a1v = *(const float4*)&rep1[(size_t)sg * DD + lane * 4];
            const float4 a2v = *(const float4*)&rep2[(size_t)sg * DD + lane * 4];
            float4 m;
            m.x = (s0 * a0.x + s1 * a1v.x + s2 * a2v.x) * INV;
            m.y = (s0 * a0.y + s1 * a1v.y + s2 * a2v.y) * INV;
            m.z = (s0 * a0.z + s1 * a1v.z + s2 * a2v.z) * INV;
            m.w = (s0 * a0.w + s1 * a1v.w + s2 * a2v.w) * INV;
            float* pp = &probes[(size_t)dg * DD + lane * 4];
            unsafeAtomicAdd(pp + 0, m.x);
            unsafeAtomicAdd(pp + 1, m.y);
            unsafeAtomicAdd(pp + 2, m.z);
            unsafeAtomicAdd(pp + 3, m.w);
        }
    }
}

__global__ void __launch_bounds__(256) wout_kernel(
    const float* __restrict__ probes, const float* __restrict__ w_out,
    float* __restrict__ out)
{
    int gw   = (blockIdx.x * 256 + threadIdx.x) >> 5;  // one wave per probe row
    int lane = threadIdx.x & 31;
    if (gw >= PTOT) return;
    const float4 v = *(const float4*)&probes[(size_t)gw * DD + lane * 4];
    const float4 w = *(const float4*)&w_out[lane * 4];
    float p = v.x * w.x + v.y * w.y + v.z * w.z + v.w * w.w;
    #pragma unroll
    for (int off = 16; off > 0; off >>= 1) p += __shfl_down(p, off, 32);
    if (lane == 0) out[gw] = p;
}

extern "C" void kernel_launch(void* const* d_in, const int* in_sizes, int n_in,
                              void* d_out, int out_size, void* d_ws, size_t ws_size,
                              hipStream_t stream) {
    (void)in_sizes; (void)n_in; (void)out_size; (void)d_ws; (void)ws_size;

    const float* atom_xyz    = (const float*)d_in[0];
    const float* probe_xyz   = (const float*)d_in[1];
    const int*   probe_edges = (const int*)  d_in[2];
    const float* ped         = (const float*)d_in[3];
    const float* cell        = (const float*)d_in[4];
    // d_in[5..7]: num_nodes/num_probes/num_probe_edges (constant = full counts)
    const float* atom_rep    = (const float*)d_in[8];
    const float* w1          = (const float*)d_in[9];
    const float* w2          = (const float*)d_in[10];
    const float* w_out       = (const float*)d_in[11];

    float* out    = (float*)d_out;
    float* probes = out + PTOT;

    // zero the whole output (out + probes accumulators)
    long total = (long)PTOT + (long)PTOT * DD;
    zero_kernel<<<2048, 256, 0, stream>>>(out, total);

    // edge kernel: 4 waves/block, 16 edges/wave
    int ntiles  = (EDGETOT + 15) / 16;    // 31250
    int nblocks = (ntiles + 3) / 4;       // 7813
    edge_kernel<<<nblocks, 128, 0, stream>>>(
        atom_xyz, probe_xyz, probe_edges, ped, cell, atom_rep, w1, w2, probes);

    // out = probes @ w_out (one wave per probe row)
    int wblocks = (PTOT * 32 + 255) / 256; // 25000
    wout_kernel<<<wblocks, 256, 0, stream>>>(probes, w_out, out);
}
